// SimpleRNN_76433238000138
// MI455X (gfx1250) — matrix operand order repaired and checked by reading, not compiled, not run on hardware
//
#include <hip/hip_runtime.h>

#define ISZ 256
#define HID 256
#define BATCH 64
#define SEQ 2048

typedef __attribute__((ext_vector_type(16))) _Float16     v16h;
typedef __attribute__((ext_vector_type(8)))  float        v8f;
typedef __attribute__((ext_vector_type(4)))  unsigned int v4u;
typedef __attribute__((ext_vector_type(4)))  float        v4f;

union Frag16 { v16h h; v4u q[2]; };

// ---------------------------------------------------------------------------
// Kernel 1: convert W (256 x 512 f32) into f16 Wx [n][k] and Wh [n][k] in d_ws.
// Row-major n-major layout matches the per-lane B-fragment load pattern below.
// ---------------------------------------------------------------------------
__global__ void rnn_cvt_w(const float* __restrict__ W,
                          _Float16* __restrict__ wx,
                          _Float16* __restrict__ wh) {
  int i = blockIdx.x * blockDim.x + threadIdx.x;   // 0 .. 131071
  int n = i >> 9;
  int c = i & 511;
  float v = W[i];
  if (c < ISZ) wx[n * ISZ + c] = (_Float16)v;
  else         wh[n * HID + (c - ISZ)] = (_Float16)v;
}

// ---------------------------------------------------------------------------
// Kernel 2: x_proj = x @ Wx^T + b, written in place into d_out[b,s,h].
// Workgroup = 256 threads = 8 waves: 2 M-halves x 4 N-quarters -> 32M x 256N.
// Each wave: 16x64 tile, 8 K-chunks of 32 -> 32 WMMAs (f16 in, f32 accum).
// ---------------------------------------------------------------------------
__global__ void __launch_bounds__(256) rnn_xproj(const float* __restrict__ x,
                                                 const _Float16* __restrict__ wx,
                                                 const float* __restrict__ bias,
                                                 float* __restrict__ out) {
  const int tid  = threadIdx.x;
  const int w    = tid >> 5, lane = tid & 31;
  const int half = lane >> 4, ml = lane & 15;
  const int nq   = w & 3, mh = w >> 2;
  const long wave_m = (long)blockIdx.x * 32 + mh * 16;
  const int  wave_n = nq * 64;

  const float* xrow = x + (wave_m + ml) * (long)ISZ;
  __builtin_prefetch(wx + (long)wave_n * ISZ, 0, 3);   // warm weight tile

  const v8f zero = {0.f,0.f,0.f,0.f,0.f,0.f,0.f,0.f};
  v8f acc[4];
#pragma unroll
  for (int nt = 0; nt < 4; ++nt) acc[nt] = zero;

#pragma unroll
  for (int kc = 0; kc < 8; ++kc) {
    // A fragment: rows of x, two 8-float runs, converted f32 -> f16.
    const int k0 = kc * 32 + half * 8;
    v4f a0 = *(const v4f*)(xrow + k0);
    v4f a1 = *(const v4f*)(xrow + k0 + 4);
    v4f a2 = *(const v4f*)(xrow + k0 + 16);
    v4f a3 = *(const v4f*)(xrow + k0 + 20);
    v16h A;
#pragma unroll
    for (int j = 0; j < 4; ++j) {
      A[j]      = (_Float16)a0[j];
      A[4 + j]  = (_Float16)a1[j];
      A[8 + j]  = (_Float16)a2[j];
      A[12 + j] = (_Float16)a3[j];
    }
#pragma unroll
    for (int nt = 0; nt < 4; ++nt) {
      const int n = wave_n + nt * 16 + ml;
      Frag16 B;
      const _Float16* bp = wx + (long)n * ISZ + kc * 32 + half * 8;
      B.q[0] = *(const v4u*)(bp);
      B.q[1] = *(const v4u*)(bp + 16);
      acc[nt] = __builtin_amdgcn_wmma_f32_16x16x32_f16(
          false, A, false, B.h, (short)0, acc[nt], false, false);
    }
  }

#pragma unroll
  for (int nt = 0; nt < 4; ++nt) {
    const int n  = wave_n + nt * 16 + ml;
    const float bv = bias[n];
#pragma unroll
    for (int r = 0; r < 8; ++r) {
      const long m = wave_m + r + 8 * half;
      out[m * (long)HID + n] = acc[nt][r] + bv;
    }
  }
}

// ---------------------------------------------------------------------------
// Kernel 3: sequential scan. 4 workgroups x 16 batch rows. Wh fragments are
// hoisted into 128 VGPRs/lane (constant across 2048 steps); hidden state lives
// in LDS (row stride 264 f16 -> conflict-free b128 fragment loads). xp is read
// from d_out and overwritten in place with sigmoid(xp + h Wh^T).
// ---------------------------------------------------------------------------
__global__ void __launch_bounds__(256, 1) rnn_scan(const _Float16* __restrict__ wh,
                                                   const float* __restrict__ h0,
                                                   float* __restrict__ out,
                                                   float* __restrict__ hlast) {
  __shared__ _Float16 Hs[16 * 264];
  const int tid  = threadIdx.x;
  const int w    = tid >> 5, lane = tid & 31;
  const int half = lane >> 4, ml = lane & 15;
  const int wgb  = blockIdx.x * 16;             // batch-row base

  // init hidden state tile from h0 (f32 -> f16)
  for (int i = tid; i < 16 * HID; i += 256) {
    int m = i >> 8, n = i & 255;
    Hs[m * 264 + n] = (_Float16)h0[(wgb + m) * HID + n];
  }

  // hoist Wh B-fragments into registers: wave owns N in [w*32, w*32+32)
  v16h Bf[2][8];
#pragma unroll
  for (int nt = 0; nt < 2; ++nt) {
    const int n = w * 32 + nt * 16 + ml;
#pragma unroll
    for (int kc = 0; kc < 8; ++kc) {
      Frag16 B;
      const _Float16* bp = wh + (long)n * HID + kc * 32 + half * 8;
      B.q[0] = *(const v4u*)(bp);
      B.q[1] = *(const v4u*)(bp + 16);
      Bf[nt][kc] = B.h;
    }
  }

  // per-(nt,r) element offsets into d_out, advanced by HID each timestep
  unsigned off[2][8];
#pragma unroll
  for (int nt = 0; nt < 2; ++nt) {
    const int n = w * 32 + nt * 16 + ml;
#pragma unroll
    for (int r = 0; r < 8; ++r) {
      const unsigned m = (unsigned)(r + 8 * half);
      off[nt][r] = ((unsigned)(wgb + m) * (unsigned)SEQ) * (unsigned)HID + (unsigned)n;
    }
  }

  __syncthreads();

  const v8f zero = {0.f,0.f,0.f,0.f,0.f,0.f,0.f,0.f};
  for (int t = 0; t < SEQ; ++t) {
    // issue xp loads first; they overlap the WMMA chain below
    float xp[2][8];
#pragma unroll
    for (int nt = 0; nt < 2; ++nt)
#pragma unroll
      for (int r = 0; r < 8; ++r)
        xp[nt][r] = out[off[nt][r]];

    v8f acc[2];
    acc[0] = zero; acc[1] = zero;
#pragma unroll
    for (int kc = 0; kc < 8; ++kc) {
      Frag16 A;
      const _Float16* ap = &Hs[ml * 264 + kc * 32 + half * 8];
      A.q[0] = *(const v4u*)(ap);
      A.q[1] = *(const v4u*)(ap + 16);
      acc[0] = __builtin_amdgcn_wmma_f32_16x16x32_f16(
          false, A.h, false, Bf[0][kc], (short)0, acc[0], false, false);
      acc[1] = __builtin_amdgcn_wmma_f32_16x16x32_f16(
          false, A.h, false, Bf[1][kc], (short)0, acc[1], false, false);
    }
    __syncthreads();   // all waves done reading Hs before it is overwritten

#pragma unroll
    for (int nt = 0; nt < 2; ++nt) {
      const int n = w * 32 + nt * 16 + ml;
#pragma unroll
      for (int r = 0; r < 8; ++r) {
        const int m = r + 8 * half;
        const float v    = acc[nt][r] + xp[nt][r];
        const float hval = 1.0f / (1.0f + __expf(-v));
        out[off[nt][r]]  = hval;              // overwrite xp in place
        Hs[m * 264 + n]  = (_Float16)hval;    // next-step A operand
        if (t == SEQ - 1) hlast[(wgb + m) * HID + n] = hval;
        off[nt][r] += HID;
      }
    }
    __syncthreads();   // Hs writes visible before next step's reads
  }
}

// ---------------------------------------------------------------------------
extern "C" void kernel_launch(void* const* d_in, const int* in_sizes, int n_in,
                              void* d_out, int out_size, void* d_ws, size_t ws_size,
                              hipStream_t stream) {
  const float* x  = (const float*)d_in[0];   // (64, 2048, 256)
  const float* h0 = (const float*)d_in[1];   // (64, 256)
  const float* W  = (const float*)d_in[2];   // (256, 512)
  const float* b  = (const float*)d_in[3];   // (256,)
  float* out = (float*)d_out;                // (64,2048,256) ++ (64,256)

  _Float16* wx = (_Float16*)d_ws;                          // 128 KiB
  _Float16* wh = (_Float16*)((char*)d_ws + ISZ * HID * 2); // 128 KiB
  float* hlast = out + (size_t)BATCH * SEQ * HID;

  rnn_cvt_w<<<(ISZ * 512) / 256, 256, 0, stream>>>(W, wx, wh);
  rnn_xproj<<<(BATCH * SEQ) / 32, 256, 0, stream>>>(x, wx, b, out);
  rnn_scan<<<BATCH / 16, 256, 0, stream>>>(wh, h0, out, hlast);
}
